// MultiHeadAttention_74577812128539
// MI455X (gfx1250) — matrix operand order repaired and checked by reading, compiled, not run on hardware
//
#include <hip/hip_runtime.h>
#include <hip/hip_bf16.h>

// ---------------------------------------------------------------------------
// MultiHeadAttention for MI455X (gfx1250, wave32, WMMA 16x16x32 f16)
// B=4, H=16, S=2048, PHD=64, DM=1024, QK_IN=128
//
// Pipeline:
//   0) prep:   Wq/Wk/Wv -> f16 transposed [H,64,IND]; Wo -> f16 (same layout)
//   1) proj:   Q (pre-scaled by 1/sqrt(128)), K row-major f16 [S,64];
//              V transposed f16 [64,S]
//   2) attn:   flash attention, scores computed transposed (K is the A
//              operand -> all WMMA operands are contiguous b128 loads);
//              causal mask applied only in the single diagonal block
//   3) outproj: OUT = O @ Wo^T + bo, f32 result
// ---------------------------------------------------------------------------

#define B_    4
#define H_    16
#define S_    2048
#define PHD_  64
#define DM_   1024
#define QKIN_ 128
#define SCALE_ 0.08838834764831845f   // 1/sqrt(128)

typedef __attribute__((ext_vector_type(16))) _Float16 v16h;
typedef __attribute__((ext_vector_type(8)))  _Float16 v8h;
typedef __attribute__((ext_vector_type(8)))  float    v8fl;
typedef __attribute__((ext_vector_type(8)))  float    v8f;

__device__ __forceinline__ v8f wmma_f16(v16h a, v16h b, v8f c) {
  return __builtin_amdgcn_wmma_f32_16x16x32_f16(false, a, false, b, (short)0, c,
                                                false, false);
}

// A-fragment (16x32 f16) from row-major f16 source, row stride ld (mult of 8).
// ISA 7.12.2: lane holds two contiguous 8-half runs of its row -> 2x b128.
__device__ __forceinline__ v16h a_frag_f16(const _Float16* p, int ld) {
  const int lane = threadIdx.x & 31;
  const int half = lane >> 4, row = lane & 15;
  const _Float16* base = p + (size_t)row * ld + half * 8;
  union { v16h v; v8h h[2]; } u;
  u.h[0] = *(const v8h*)(base);        // K = half*8 + 0..7
  u.h[1] = *(const v8h*)(base + 16);   // K = 16 + half*8 + 0..7
  return u.v;
}

// A-fragment from row-major f32 source (convert inline, vector loads).
__device__ __forceinline__ v16h a_frag_f32(const float* p, int ld) {
  const int lane = threadIdx.x & 31;
  const int half = lane >> 4, row = lane & 15;
  const float* base = p + (size_t)row * ld + half * 8;
  v8fl lo = *(const v8fl*)(base);
  v8fl hi = *(const v8fl*)(base + 16);
  v16h a;
#pragma unroll
  for (int i = 0; i < 8; ++i) { a[i] = (_Float16)lo[i]; a[8 + i] = (_Float16)hi[i]; }
  return a;
}

// B-fragment (32x16 f16) from an N-major source: element(kk,n) = pT[n*ld+kk].
// Lane n reads K = half*8 + 0..7 and 16 + half*8 + 0..7 -> 2x b128.
__device__ __forceinline__ v16h b_frag_f16T(const _Float16* pT, int ld) {
  const int lane = threadIdx.x & 31;
  const int n = lane & 15;
  const int koff = (lane >> 4) * 8;
  const _Float16* base = pT + (size_t)n * ld + koff;
  union { v16h v; v8h h[2]; } u;
  u.h[0] = *(const v8h*)(base);
  u.h[1] = *(const v8h*)(base + 16);
  return u.v;
}

// ---------------------------------------------------------------------------
// Kernel 0: weight prep. Wq/Wk/Wv [H,IND,64] f32 -> transposed f16 [H,64,IND];
// Wo [DM,DM] f32 -> f16 same layout (N-major for x @ Wo^T already).
// ---------------------------------------------------------------------------
__global__ void __launch_bounds__(256) mha_prep_weights(
    const float* __restrict__ Wq, const float* __restrict__ Wk,
    const float* __restrict__ Wv, const float* __restrict__ Wo,
    _Float16* __restrict__ WqT, _Float16* __restrict__ WkT,
    _Float16* __restrict__ WvT, _Float16* __restrict__ Wo16)
{
  const int NQK = H_ * QKIN_ * PHD_;   // 131072
  const int NV  = H_ * PHD_ * PHD_;    // 65536
  const int NO  = DM_ * DM_;           // 1048576
  const int stride = (int)(gridDim.x * blockDim.x);
  for (int i = (int)(blockIdx.x * blockDim.x + threadIdx.x); i < NO; i += stride) {
    Wo16[i] = (_Float16)Wo[i];
    if (i < NQK) {
      const int h = i / (QKIN_ * PHD_);
      const int r = i - h * (QKIN_ * PHD_);
      const int d = r / PHD_;          // input feature
      const int n = r - d * PHD_;      // output feature
      const size_t to = (size_t)h * PHD_ * QKIN_ + (size_t)n * QKIN_ + d;
      WqT[to] = (_Float16)Wq[i];
      WkT[to] = (_Float16)Wk[i];
    }
    if (i < NV) {
      const int h = i / (PHD_ * PHD_);
      const int r = i - h * (PHD_ * PHD_);
      const int d = r / PHD_;
      const int n = r - d * PHD_;
      WvT[(size_t)h * PHD_ * PHD_ + (size_t)n * PHD_ + d] = (_Float16)Wv[i];
    }
  }
}

// ---------------------------------------------------------------------------
// Kernel 1: fused Q/K/V projections. One wave per 16x64 output tile.
// Q (x SCALE) and K stored row-major f16 [S,64]; V stored transposed [64,S].
// ---------------------------------------------------------------------------
__global__ void __launch_bounds__(256) mha_proj_qkv(
    const float* __restrict__ q, const float* __restrict__ k,
    const float* __restrict__ v,
    const _Float16* __restrict__ WqT, const float* __restrict__ bq,
    const _Float16* __restrict__ WkT, const float* __restrict__ bk,
    const _Float16* __restrict__ WvT, const float* __restrict__ bv,
    _Float16* __restrict__ Qh, _Float16* __restrict__ Kh,
    _Float16* __restrict__ Vt)
{
  const int wave = (int)((blockIdx.x * blockDim.x + threadIdx.x) >> 5);
  const int proj = wave >> 13;          // 0=Q, 1=K, 2=V
  const int rem  = wave & 8191;
  const int bh   = rem >> 7;
  const int s0   = (rem & 127) * 16;
  const int h    = bh & (H_ - 1);

  const float* in; const _Float16* WT; const float* bias; int IND;
  if (proj == 0)      { in = q + (size_t)bh * S_ * QKIN_; WT = WqT + (size_t)h * PHD_ * QKIN_; bias = bq + h * PHD_; IND = QKIN_; }
  else if (proj == 1) { in = k + (size_t)bh * S_ * QKIN_; WT = WkT + (size_t)h * PHD_ * QKIN_; bias = bk + h * PHD_; IND = QKIN_; }
  else                { in = v + (size_t)bh * S_ * PHD_;  WT = WvT + (size_t)h * PHD_ * PHD_;  bias = bv + h * PHD_; IND = PHD_; }

  v8f acc[4];
#pragma unroll
  for (int nt = 0; nt < 4; ++nt)
#pragma unroll
    for (int r = 0; r < 8; ++r) acc[nt][r] = 0.0f;

  for (int kb = 0; kb < IND; kb += 32) {
    v16h A = a_frag_f32(in + (size_t)s0 * IND + kb, IND);
#pragma unroll
    for (int nt = 0; nt < 4; ++nt) {
      v16h Bf = b_frag_f16T(WT + (size_t)(nt * 16) * IND + kb, IND);
      acc[nt] = wmma_f16(A, Bf, acc[nt]);
    }
  }

  const int lane = threadIdx.x & 31;
  const int half = lane >> 4, nl = lane & 15;
  // fold the attention scale into Q so the flash inner loop skips it
  const float outscale = (proj == 0) ? SCALE_ : 1.0f;
  if (proj < 2) {
    _Float16* out = (proj == 0 ? Qh : Kh) + (size_t)bh * S_ * PHD_;
#pragma unroll
    for (int nt = 0; nt < 4; ++nt) {
      const float bb = bias[nt * 16 + nl];
#pragma unroll
      for (int r = 0; r < 8; ++r)
        out[(size_t)(s0 + r + half * 8) * PHD_ + nt * 16 + nl] =
            (_Float16)((acc[nt][r] + bb) * outscale);
    }
  } else {
    _Float16* out = Vt + (size_t)bh * S_ * PHD_;   // [PHD][S] layout
#pragma unroll
    for (int nt = 0; nt < 4; ++nt) {
      const float bb = bias[nt * 16 + nl];
#pragma unroll
      for (int r = 0; r < 8; ++r)
        out[(size_t)(nt * 16 + nl) * S_ + s0 + r + half * 8] =
            (_Float16)(acc[nt][r] + bb);
    }
  }
}

// ---------------------------------------------------------------------------
// Kernel 2: causal flash attention, one wave per 16-query tile.
// scores^T = K (A, streamed, b128 fast path) @ Q^T (B, loop-invariant).
// Softmax stats are per-lane scalars (query = column). P^T -> P via LDS
// (2x b128 store + 2x b128 load). O += P @ V with V transposed (fast path).
// Main loop is unmasked; exactly one diagonal block needs the causal mask.
// ---------------------------------------------------------------------------
#define ATT_WAVES 8

struct AttState {
  float mprev, lsum;
  v8f acc[4];
};

__device__ __forceinline__ void attn_block(
    const _Float16* __restrict__ Kp, const _Float16* __restrict__ Vtp,
    _Float16* pbuf, v16h QB0, v16h QB1, int kb, int q0, bool masked,
    AttState& st_)
{
  const int lane = threadIdx.x & 31;
  const int half = lane >> 4, nl = lane & 15;

  // scores^T: two 16-key tiles (M) x 16 queries (N)
  v8f st[2];
#pragma unroll
  for (int kt = 0; kt < 2; ++kt) {
    const _Float16* Kbase = Kp + (size_t)(kb + kt * 16) * PHD_;
    v16h KA0 = a_frag_f16(Kbase, PHD_);
    v16h KA1 = a_frag_f16(Kbase + 32, PHD_);
    v8f c;
#pragma unroll
    for (int r = 0; r < 8; ++r) c[r] = 0.0f;
    c = wmma_f16(KA0, QB0, c);
    c = wmma_f16(KA1, QB1, c);
    st[kt] = c;
  }

  if (masked) {
    const int myq = q0 + nl;
#pragma unroll
    for (int kt = 0; kt < 2; ++kt)
#pragma unroll
      for (int r = 0; r < 8; ++r) {
        const int key = kb + kt * 16 + r + half * 8;
        st[kt][r] = (key <= myq) ? st[kt][r] : -1.0e30f;
      }
  }

  float lmax = -1.0e30f;
#pragma unroll
  for (int kt = 0; kt < 2; ++kt)
#pragma unroll
    for (int r = 0; r < 8; ++r) lmax = fmaxf(lmax, st[kt][r]);
  lmax = fmaxf(lmax, __shfl_xor(lmax, 16, 32));   // other half, same query
  const float mn = fmaxf(st_.mprev, lmax);
  const float alpha = __expf(st_.mprev - mn);
  float ls = 0.0f;
#pragma unroll
  for (int kt = 0; kt < 2; ++kt)
#pragma unroll
    for (int r = 0; r < 8; ++r) {
      const float p = __expf(st[kt][r] - mn);
      st[kt][r] = p;
      ls += p;
    }
  ls += __shfl_xor(ls, 16, 32);
  st_.lsum = st_.lsum * alpha + ls;
  st_.mprev = mn;

  // rescale O accumulators: acc row query = r + 8*half -> fetch that
  // query's alpha from the lane that owns it
#pragma unroll
  for (int r = 0; r < 8; ++r) {
    const float ar = __shfl(alpha, r + half * 8, 32);
#pragma unroll
    for (int nt = 0; nt < 4; ++nt) st_.acc[nt][r] *= ar;
  }

  // P^T (D layout) -> LDS row-major P [query][key] via packed b128 stores
#pragma unroll
  for (int kt = 0; kt < 2; ++kt) {
    v8h pk;
#pragma unroll
    for (int r = 0; r < 8; ++r) pk[r] = (_Float16)st[kt][r];
    *(v8h*)(pbuf + nl * 32 + kt * 16 + half * 8) = pk;
  }
  asm volatile("s_wait_dscnt 0" ::: "memory");
  v16h PA = a_frag_f16(pbuf, 32);

  // O += P (16x32) @ V (32x64); V is [PHD][S] -> fast-path B frags
#pragma unroll
  for (int nt = 0; nt < 4; ++nt) {
    v16h BV = b_frag_f16T(Vtp + (size_t)(nt * 16) * S_ + kb, S_);
    st_.acc[nt] = wmma_f16(PA, BV, st_.acc[nt]);
  }
}

__global__ void __launch_bounds__(256) mha_flash_attn(
    const _Float16* __restrict__ Qh, const _Float16* __restrict__ Kh,
    const _Float16* __restrict__ Vt, _Float16* __restrict__ Ows)
{
  __shared__ _Float16 lds_p[ATT_WAVES * 16 * 32];
  _Float16* pbuf = &lds_p[(threadIdx.x >> 5) * 16 * 32];

  const int wave = (int)((blockIdx.x * blockDim.x + threadIdx.x) >> 5);
  const int bh = wave >> 7;
  const int q0 = (wave & 127) * 16;
  const int h  = bh & (H_ - 1);
  const int b  = bh >> 4;

  const _Float16* Qp  = Qh + ((size_t)bh * S_ + q0) * PHD_;
  const _Float16* Kp  = Kh + (size_t)bh * S_ * PHD_;
  const _Float16* Vtp = Vt + (size_t)bh * S_ * PHD_;   // [PHD][S]

  const int lane = threadIdx.x & 31;
  const int half = lane >> 4, nl = lane & 15;

  // Q^T B-fragments (feature chunks 0..31, 32..63), loop-invariant
  v16h QB0 = b_frag_f16T(Qp, PHD_);
  v16h QB1 = b_frag_f16T(Qp + 32, PHD_);

  AttState st_;
  st_.mprev = -1.0e30f;
  st_.lsum  = 0.0f;
#pragma unroll
  for (int nt = 0; nt < 4; ++nt)
#pragma unroll
    for (int r = 0; r < 8; ++r) st_.acc[nt][r] = 0.0f;

  // blocks [0, kfull): fully below the diagonal, no mask needed;
  // block at kfull: the single partially-masked diagonal block.
  const int kfull = q0 & ~31;
  for (int kb = 0; kb < kfull; kb += 32) {
    __builtin_prefetch(Kp + (size_t)(kb + 32) * PHD_, 0, 3);
    __builtin_prefetch(Vtp + kb + 32, 0, 3);
    attn_block(Kp, Vtp, pbuf, QB0, QB1, kb, q0, false, st_);
  }
  attn_block(Kp, Vtp, pbuf, QB0, QB1, kfull, q0, true, st_);

  // normalize and store f16 in [B, S, H*PHD]
  _Float16* Op = Ows + ((size_t)b * S_ + q0) * DM_ + h * PHD_;
#pragma unroll
  for (int r = 0; r < 8; ++r) {
    const float ln = __shfl(st_.lsum, r + half * 8, 32);
    const float inv = 1.0f / ln;
#pragma unroll
    for (int nt = 0; nt < 4; ++nt)
      Op[(size_t)(r + half * 8) * DM_ + nt * 16 + nl] =
          (_Float16)(st_.acc[nt][r] * inv);
  }
}

// ---------------------------------------------------------------------------
// Kernel 3: OUT[m,n] = sum_k O[m,k] * Wo16[n,k] + bo[n]   (x @ Wo^T)
// Both operands on the b128 fast path. One wave per 16x64 tile.
// ---------------------------------------------------------------------------
__global__ void __launch_bounds__(256) mha_out_proj(
    const _Float16* __restrict__ Ows, const _Float16* __restrict__ Wo16,
    const float* __restrict__ bo, float* __restrict__ out)
{
  const int wave = (int)((blockIdx.x * blockDim.x + threadIdx.x) >> 5);
  const int m0 = (wave >> 4) * 16;
  const int n0 = (wave & 15) * 64;

  v8f acc[4];
#pragma unroll
  for (int nt = 0; nt < 4; ++nt)
#pragma unroll
    for (int r = 0; r < 8; ++r) acc[nt][r] = 0.0f;

  for (int kb = 0; kb < DM_; kb += 32) {
    v16h A = a_frag_f16(Ows + (size_t)m0 * DM_ + kb, DM_);
#pragma unroll
    for (int nt = 0; nt < 4; ++nt) {
      v16h Bf = b_frag_f16T(Wo16 + (size_t)(n0 + nt * 16) * DM_ + kb, DM_);
      acc[nt] = wmma_f16(A, Bf, acc[nt]);
    }
  }

  const int lane = threadIdx.x & 31;
  const int half = lane >> 4, nl = lane & 15;
#pragma unroll
  for (int nt = 0; nt < 4; ++nt) {
    const float bb = bo[n0 + nt * 16 + nl];
#pragma unroll
    for (int r = 0; r < 8; ++r)
      out[(size_t)(m0 + r + half * 8) * DM_ + n0 + nt * 16 + nl] =
          acc[nt][r] + bb;
  }
}

// ---------------------------------------------------------------------------
extern "C" void kernel_launch(void* const* d_in, const int* in_sizes, int n_in,
                              void* d_out, int out_size, void* d_ws, size_t ws_size,
                              hipStream_t stream) {
  const float* q  = (const float*)d_in[0];
  const float* k  = (const float*)d_in[1];
  const float* v  = (const float*)d_in[2];
  const float* Wq = (const float*)d_in[3];
  const float* bq = (const float*)d_in[4];
  const float* Wk = (const float*)d_in[5];
  const float* bk = (const float*)d_in[6];
  const float* Wv = (const float*)d_in[7];
  const float* bv = (const float*)d_in[8];
  const float* Wo = (const float*)d_in[9];
  const float* bo = (const float*)d_in[10];
  // d_in[11] = mask (causality computed analytically)
  float* out = (float*)d_out;

  const size_t panel = (size_t)B_ * H_ * S_ * PHD_;   // 8Mi halves
  _Float16* Qh   = (_Float16*)d_ws;
  _Float16* Kh   = Qh + panel;
  _Float16* Vt   = Kh + panel;
  _Float16* Ows  = Vt + panel;
  _Float16* WqT  = Ows + panel;
  _Float16* WkT  = WqT + (size_t)H_ * PHD_ * QKIN_;
  _Float16* WvT  = WkT + (size_t)H_ * PHD_ * QKIN_;
  _Float16* Wo16 = WvT + (size_t)H_ * PHD_ * PHD_;

  mha_prep_weights<<<512, 256, 0, stream>>>(Wq, Wk, Wv, Wo, WqT, WkT, WvT, Wo16);
  mha_proj_qkv<<<3072, 256, 0, stream>>>(q, k, v, WqT, bq, WkT, bk, WvT, bv,
                                         Qh, Kh, Vt);
  mha_flash_attn<<<1024, 256, 0, stream>>>(Qh, Kh, Vt, Ows);
  mha_out_proj<<<1024, 256, 0, stream>>>(Ows, Wo16, bo, out);
}
